// BertModel_68856915690236
// MI455X (gfx1250) — matrix operand order repaired and checked by reading, not compiled, not run on hardware
//
#include <hip/hip_runtime.h>
#include <hip/hip_bf16.h>
#include <math.h>

#define BB   4
#define SS   2048
#define HH   768
#define NHH  12
#define DHH  64
#define FFF  3072
#define LL   12
#define MTOT (BB * SS) /* 8192 */

typedef __bf16 bf16_t;
typedef bf16_t v16bf __attribute__((ext_vector_type(16)));
typedef float  v8f   __attribute__((ext_vector_type(8)));

union FragAB { v16bf v; unsigned short s[16]; };

__device__ __forceinline__ unsigned short f2bf(float f) {
  union { float f; unsigned u; } x; x.f = f;
  unsigned r = x.u + 0x7FFFu + ((x.u >> 16) & 1u);  // round-to-nearest-even
  return (unsigned short)(r >> 16);
}

// ---- async global->LDS copy path (guarded; falls back to plain copies) ----
#if defined(__has_builtin)
#if __has_builtin(__builtin_amdgcn_global_load_async_to_lds_b64) && \
    __has_builtin(__builtin_amdgcn_s_wait_asynccnt)
#define HAVE_ASYNC 1
#endif
#endif
#ifndef HAVE_ASYNC
#define HAVE_ASYNC 0
#endif

#if HAVE_ASYNC
typedef int v2i_t __attribute__((vector_size(8)));
typedef __attribute__((address_space(1))) v2i_t g_v2i;
typedef __attribute__((address_space(3))) v2i_t l_v2i;
#endif

__device__ __forceinline__ void async_copy_b64(const void* g, void* l) {
#if HAVE_ASYNC
  __builtin_amdgcn_global_load_async_to_lds_b64(
      (g_v2i*)(size_t)g, (l_v2i*)(unsigned)(size_t)l, 0, 0);
#else
  *(unsigned long long*)l = *(const unsigned long long*)g;
#endif
}

#if HAVE_ASYNC
#define ASYNC_WAIT(n) __builtin_amdgcn_s_wait_asynccnt(n)
#else
#define ASYNC_WAIT(n) do {} while (0)
#endif

// ---------------------------------------------------------------------------
// Weight convert + transpose: W[K,N] fp32 -> WT[N,K] bf16 (tiled via LDS)
// grid (N/32, K/32), block (32,8)
// ---------------------------------------------------------------------------
__global__ __launch_bounds__(256)
void cvt_wT(const float* __restrict__ W, unsigned short* __restrict__ WT,
            int K, int N)
{
  __shared__ float t[32][33];
  const int n0 = blockIdx.x * 32, k0 = blockIdx.y * 32;
  for (int kk = threadIdx.y; kk < 32; kk += 8)
    t[kk][threadIdx.x] = W[(size_t)(k0 + kk) * N + n0 + threadIdx.x];
  __syncthreads();
  for (int nn = threadIdx.y; nn < 32; nn += 8)
    WT[(size_t)(n0 + nn) * K + k0 + threadIdx.x] = f2bf(t[threadIdx.x][nn]);
}

// ---------------------------------------------------------------------------
// bf16 WMMA GEMM: Y[M,N] = X[M,K] * WT[N,K]^T + bias (+res | +gelu)
// 128x128 block tile, K-step 32, double-buffered async LDS staging.
// 8 waves (4x2): each wave 32x64 (2x4 wmma tiles).
// mode: 0 = bias -> bf16 out, 1 = bias+residual -> fp32 out,
//       2 = bias+erf GELU -> bf16 out
// ---------------------------------------------------------------------------
__global__ __launch_bounds__(256)
void wmma_gemm(const unsigned short* __restrict__ Xb,
               const unsigned short* __restrict__ WT,
               const float* __restrict__ bias, const float* __restrict__ res,
               float* __restrict__ Yf, unsigned short* __restrict__ Yb,
               int K, int N, int mode)
{
  __shared__ unsigned short As[2][128][36];  // [m][k], 72B row (8B aligned)
  __shared__ unsigned short Bs[2][128][36];  // [n][k]

  const int tid  = threadIdx.x;
  const int lane = tid & 31;
  const int w    = tid >> 5;
  const int wm   = w & 3, wn = w >> 2;
  const int lh   = lane >> 4, lc = lane & 15;
  const int m0   = blockIdx.y * 128, n0 = blockIdx.x * 128;

  v8f zero = {};
  v8f acc[2][4];
  for (int mt = 0; mt < 2; ++mt)
    for (int nt = 0; nt < 4; ++nt) acc[mt][nt] = zero;

  const int nk = K >> 5;

  // stage(buf, k0): 8 async b64 copies per thread (4 A + 4 B)
  auto stage = [&](int buf, int k0) {
    #pragma unroll
    for (int t = 0; t < 4; ++t) {
      int i = tid + t * 256;
      int r = i >> 3, c4 = (i & 7) << 2;
      async_copy_b64(Xb + (size_t)(m0 + r) * K + k0 + c4, &As[buf][r][c4]);
    }
    #pragma unroll
    for (int t = 0; t < 4; ++t) {
      int i = tid + t * 256;
      int r = i >> 3, c4 = (i & 7) << 2;
      async_copy_b64(WT + (size_t)(n0 + r) * K + k0 + c4, &Bs[buf][r][c4]);
    }
  };

  stage(0, 0);
  for (int kt = 0; kt < nk; ++kt) {
    const int cur = kt & 1;
    if (kt + 1 < nk) { stage(cur ^ 1, (kt + 1) << 5); ASYNC_WAIT(8); }
    else             { ASYNC_WAIT(0); }
    __syncthreads();

    FragAB a[2];
    for (int mt = 0; mt < 2; ++mt) {
      int rr = wm * 32 + mt * 16 + lc;
      #pragma unroll
      for (int j = 0; j < 8; ++j) {
        a[mt].s[j]     = As[cur][rr][lh * 8 + j];
        a[mt].s[8 + j] = As[cur][rr][16 + lh * 8 + j];
      }
    }
    for (int nt = 0; nt < 4; ++nt) {
      FragAB bfr;
      int nn = wn * 64 + nt * 16 + lc;
      #pragma unroll
      for (int j = 0; j < 16; ++j) bfr.s[j] = Bs[cur][nn][(j & 8) * 2 + lh * 8 + (j & 7)];
      acc[0][nt] = __builtin_amdgcn_wmma_f32_16x16x32_bf16(
          false, a[0].v, false, bfr.v, (short)0, acc[0][nt], false, false);
      acc[1][nt] = __builtin_amdgcn_wmma_f32_16x16x32_bf16(
          false, a[1].v, false, bfr.v, (short)0, acc[1][nt], false, false);
    }
    __syncthreads();
  }

  for (int mt = 0; mt < 2; ++mt)
    for (int nt = 0; nt < 4; ++nt) {
      #pragma unroll
      for (int r = 0; r < 8; ++r) {
        int row = m0 + wm * 32 + mt * 16 + lh * 8 + r;
        int col = n0 + wn * 64 + nt * 16 + lc;
        float val = acc[mt][nt][r] + bias[col];
        if (mode == 1) {
          val += res[(size_t)row * N + col];
          Yf[(size_t)row * N + col] = val;
        } else {
          if (mode == 2) val = 0.5f * val * (1.0f + erff(val * 0.70710678f));
          Yb[(size_t)row * N + col] = f2bf(val);
        }
      }
    }
}

// ---------------------------------------------------------------------------
// Flash attention (bf16 q/k/v in, bf16 ctx out).
// grid (S/64, NH, B), 128 threads (4 waves x 16 query rows).
// ---------------------------------------------------------------------------
__global__ __launch_bounds__(128)
void flash_attn(const unsigned short* __restrict__ Qb,
                const unsigned short* __restrict__ Kb,
                const unsigned short* __restrict__ Vb,
                const float* __restrict__ mask,
                unsigned short* __restrict__ Ob)
{
  __shared__ unsigned short Ks[64][68];      // [key][d], 136B rows (8B align)
  __shared__ unsigned short Vs[64][68];      // transposed [d][key]
  __shared__ unsigned short Ps[4][16][68];   // per-wave P tile [m][key]

  const int tid  = threadIdx.x;
  const int wv   = tid >> 5;
  const int lane = tid & 31;
  const int lh   = lane >> 4, lc = lane & 15;
  const int qb   = blockIdx.x * 64;
  const int head = blockIdx.y;
  const int b    = blockIdx.z;
  const float scale = 0.125f;  // 1/sqrt(64), applied to scores

  // Q A-fragments: raw bf16, 16B vector loads
  FragAB qa[2];
  {
    const unsigned short* qptr =
        Qb + ((size_t)(b * SS) + qb + wv * 16 + lc) * HH + head * DHH;
    #pragma unroll
    for (int st = 0; st < 2; ++st) {
      *(uint4*)&qa[st].s[0] = *(const uint4*)(qptr + st * 32 + lh * 8);
      *(uint4*)&qa[st].s[8] = *(const uint4*)(qptr + st * 32 + 16 + lh * 8);
    }
  }

  v8f zero = {};
  v8f oacc[4];
  for (int nt = 0; nt < 4; ++nt) oacc[nt] = zero;
  float mi[8], li[8];
  for (int r = 0; r < 8; ++r) { mi[r] = -3.0e38f; li[r] = 0.0f; }

  for (int kb = 0; kb < SS; kb += 64) {
    __syncthreads();
    // K block: async copies (4 bf16 per op, 8 ops/thread)
    #pragma unroll
    for (int t = 0; t < 8; ++t) {
      int i = tid + t * 128;
      int kr = i >> 4, c4 = (i & 15) << 2;
      async_copy_b64(Kb + ((size_t)(b * SS) + kb + kr) * HH + head * DHH + c4,
                     &Ks[kr][c4]);
    }
    // V block transposed: manual bf16 scatter to LDS
    for (int i = tid; i < 4096; i += 128) {
      int kr = i >> 6, d = i & 63;
      Vs[d][kr] = Vb[((size_t)(b * SS) + kb + kr) * HH + head * DHH + d];
    }
    ASYNC_WAIT(0);
    __syncthreads();

    // scores = Q * K^T
    v8f sacc[4];
    for (int nt = 0; nt < 4; ++nt) sacc[nt] = zero;
    for (int st = 0; st < 2; ++st)
      for (int nt = 0; nt < 4; ++nt) {
        FragAB kf;
        #pragma unroll
        for (int j = 0; j < 16; ++j)
          kf.s[j] = Ks[nt * 16 + lc][st * 32 + (j & 8) * 2 + lh * 8 + (j & 7)];
        sacc[nt] = __builtin_amdgcn_wmma_f32_16x16x32_bf16(
            false, qa[st].v, false, kf.v, (short)0, sacc[nt], false, false);
      }

    // scale + attention mask
    for (int nt = 0; nt < 4; ++nt) {
      float mv = mask[b * SS + kb + nt * 16 + lc];
      #pragma unroll
      for (int r = 0; r < 8; ++r) sacc[nt][r] = sacc[nt][r] * scale + mv;
    }

    // online softmax (rows live in 16-lane halves)
    for (int r = 0; r < 8; ++r) {
      float mx = fmaxf(fmaxf(sacc[0][r], sacc[1][r]),
                       fmaxf(sacc[2][r], sacc[3][r]));
      mx = fmaxf(mx, __shfl_xor(mx, 1));
      mx = fmaxf(mx, __shfl_xor(mx, 2));
      mx = fmaxf(mx, __shfl_xor(mx, 4));
      mx = fmaxf(mx, __shfl_xor(mx, 8));
      float nm = fmaxf(mi[r], mx);
      float alpha = __expf(mi[r] - nm);
      mi[r] = nm;
      float rs = 0.0f;
      for (int nt = 0; nt < 4; ++nt) {
        float p = __expf(sacc[nt][r] - nm);
        sacc[nt][r] = p;
        rs += p;
      }
      rs += __shfl_xor(rs, 1);
      rs += __shfl_xor(rs, 2);
      rs += __shfl_xor(rs, 4);
      rs += __shfl_xor(rs, 8);
      li[r] = li[r] * alpha + rs;
      for (int nt = 0; nt < 4; ++nt) oacc[nt][r] *= alpha;
    }

    // C-layout -> LDS bf16 -> A-layout for P*V
    for (int nt = 0; nt < 4; ++nt)
      #pragma unroll
      for (int r = 0; r < 8; ++r)
        Ps[wv][lh * 8 + r][nt * 16 + lc] = f2bf(sacc[nt][r]);
    __syncthreads();

    for (int st = 0; st < 2; ++st) {
      FragAB pa;
      #pragma unroll
      for (int j = 0; j < 8; ++j) {
        pa.s[j]     = Ps[wv][lc][st * 32 + lh * 8 + j];
        pa.s[8 + j] = Ps[wv][lc][st * 32 + 16 + lh * 8 + j];
      }
      for (int nt = 0; nt < 4; ++nt) {
        FragAB vf;
        #pragma unroll
        for (int j = 0; j < 16; ++j)
          vf.s[j] = Vs[nt * 16 + lc][st * 32 + (j & 8) * 2 + lh * 8 + (j & 7)];
        oacc[nt] = __builtin_amdgcn_wmma_f32_16x16x32_bf16(
            false, pa.v, false, vf.v, (short)0, oacc[nt], false, false);
      }
    }
  }

  unsigned short* optr =
      Ob + ((size_t)(b * SS) + qb + wv * 16) * HH + head * DHH;
  for (int nt = 0; nt < 4; ++nt)
    #pragma unroll
    for (int r = 0; r < 8; ++r)
      optr[(size_t)(lh * 8 + r) * HH + nt * 16 + lc] = f2bf(oacc[nt][r] / li[r]);
}

// ---------------------------------------------------------------------------
// LayerNorm over rows of 768, dual fp32 + bf16 output
// ---------------------------------------------------------------------------
__global__ __launch_bounds__(256)
void layernorm_k(const float* __restrict__ X, const float* __restrict__ w,
                 const float* __restrict__ bpar, float* __restrict__ Yf,
                 unsigned short* __restrict__ Yb)
{
  __shared__ float red[256], red2[256];
  const int row = blockIdx.x;
  const int tid = threadIdx.x;
  const float* x = X + (size_t)row * HH;
  float vals[3], s = 0.f, s2 = 0.f;
  #pragma unroll
  for (int c = 0; c < 3; ++c) {
    float v = x[tid + c * 256];
    vals[c] = v; s += v; s2 += v * v;
  }
  red[tid] = s; red2[tid] = s2;
  __syncthreads();
  for (int off = 128; off > 0; off >>= 1) {
    if (tid < off) { red[tid] += red[tid + off]; red2[tid] += red2[tid + off]; }
    __syncthreads();
  }
  float mean = red[0] * (1.0f / 768.0f);
  float var  = red2[0] * (1.0f / 768.0f) - mean * mean;
  float rstd = rsqrtf(var + 1e-12f);
  #pragma unroll
  for (int c = 0; c < 3; ++c) {
    int j = tid + c * 256;
    float v = w[j] * (vals[c] - mean) * rstd + bpar[j];
    Yf[(size_t)row * HH + j] = v;
    Yb[(size_t)row * HH + j] = f2bf(v);
  }
}

// ---------------------------------------------------------------------------
// Embedding gather + LayerNorm, dual fp32 + bf16 output
// ---------------------------------------------------------------------------
__global__ __launch_bounds__(256)
void embed_ln_k(const int* __restrict__ ids, const float* __restrict__ we,
                const float* __restrict__ pe, const float* __restrict__ te,
                const float* __restrict__ w, const float* __restrict__ bpar,
                float* __restrict__ Yf, unsigned short* __restrict__ Yb)
{
  __shared__ float red[256], red2[256];
  const int row = blockIdx.x;
  const int pos = row & (SS - 1);
  const int tid = threadIdx.x;
  const int id  = ids[row];
  float vals[3], s = 0.f, s2 = 0.f;
  #pragma unroll
  for (int c = 0; c < 3; ++c) {
    int j = tid + c * 256;
    float v = we[(size_t)id * HH + j] + pe[(size_t)pos * HH + j] + te[j];
    vals[c] = v; s += v; s2 += v * v;
  }
  red[tid] = s; red2[tid] = s2;
  __syncthreads();
  for (int off = 128; off > 0; off >>= 1) {
    if (tid < off) { red[tid] += red[tid + off]; red2[tid] += red2[tid + off]; }
    __syncthreads();
  }
  float mean = red[0] * (1.0f / 768.0f);
  float var  = red2[0] * (1.0f / 768.0f) - mean * mean;
  float rstd = rsqrtf(var + 1e-12f);
  #pragma unroll
  for (int c = 0; c < 3; ++c) {
    int j = tid + c * 256;
    float v = w[j] * (vals[c] - mean) * rstd + bpar[j];
    Yf[(size_t)row * HH + j] = v;
    Yb[(size_t)row * HH + j] = f2bf(v);
  }
}

// ---------------------------------------------------------------------------
extern "C" void kernel_launch(void* const* d_in, const int* in_sizes, int n_in,
                              void* d_out, int out_size, void* d_ws, size_t ws_size,
                              hipStream_t stream)
{
  (void)in_sizes; (void)n_in; (void)out_size; (void)ws_size;
  const int*   input_ids = (const int*)  d_in[0];
  const float* attn_mask = (const float*)d_in[1];
  const float* word_emb  = (const float*)d_in[2];
  const float* pos_emb   = (const float*)d_in[3];
  const float* type_emb  = (const float*)d_in[4];
  const float* ln_e_w    = (const float*)d_in[5];
  const float* ln_e_b    = (const float*)d_in[6];
  const float* Wq  = (const float*)d_in[7];
  const float* bq  = (const float*)d_in[8];
  const float* Wk  = (const float*)d_in[9];
  const float* bk  = (const float*)d_in[10];
  const float* Wv  = (const float*)d_in[11];
  const float* bv  = (const float*)d_in[12];
  const float* Wo  = (const float*)d_in[13];
  const float* bo  = (const float*)d_in[14];
  const float* ln1w = (const float*)d_in[15];
  const float* ln1b = (const float*)d_in[16];
  const float* Wi  = (const float*)d_in[17];
  const float* bi  = (const float*)d_in[18];
  const float* Wf  = (const float*)d_in[19];
  const float* bfp = (const float*)d_in[20];
  const float* ln2w = (const float*)d_in[21];
  const float* ln2b = (const float*)d_in[22];

  const size_t MH = (size_t)MTOT * HH;
  float* h    = (float*)d_ws;
  float* attn = h + MH;
  float* tmp  = attn + MH;
  unsigned short* hbf    = (unsigned short*)(tmp + MH);
  unsigned short* attnbf = hbf + MH;
  unsigned short* ctxb   = attnbf + MH;
  unsigned short* qb2    = ctxb + MH;
  unsigned short* kb2    = qb2 + MH;
  unsigned short* vb2    = kb2 + MH;
  unsigned short* ffb    = vb2 + MH;                 // M x FF bf16
  unsigned short* wqT    = ffb + (size_t)MTOT * FFF; // per-layer WT buffers
  unsigned short* wkT    = wqT + (size_t)HH * HH;
  unsigned short* wvT    = wkT + (size_t)HH * HH;
  unsigned short* woT    = wvT + (size_t)HH * HH;
  unsigned short* wiT    = woT + (size_t)HH * HH;    // [FFF, HH]
  unsigned short* wfT    = wiT + (size_t)HH * FFF;   // [HH, FFF]

  embed_ln_k<<<MTOT, 256, 0, stream>>>(input_ids, word_emb, pos_emb, type_emb,
                                       ln_e_w, ln_e_b, h, hbf);

  dim3 gH(HH / 128, MTOT / 128);    // (6, 64)
  dim3 gF(FFF / 128, MTOT / 128);   // (24, 64)
  dim3 gA(SS / 64, NHH, BB);        // (32, 12, 4)
  dim3 cb(32, 8);

  for (int l = 0; l < LL; ++l) {
    const float* Wq_l = Wq + (size_t)l * HH * HH;
    const float* Wk_l = Wk + (size_t)l * HH * HH;
    const float* Wv_l = Wv + (size_t)l * HH * HH;
    const float* Wo_l = Wo + (size_t)l * HH * HH;
    const float* Wi_l = Wi + (size_t)l * HH * FFF;
    const float* Wf_l = Wf + (size_t)l * FFF * HH;
    const float* bq_l = bq + (size_t)l * HH;
    const float* bk_l = bk + (size_t)l * HH;
    const float* bv_l = bv + (size_t)l * HH;
    const float* bo_l = bo + (size_t)l * HH;
    const float* bi_l = bi + (size_t)l * FFF;
    const float* bf_l = bfp + (size_t)l * HH;
    const float* l1w = ln1w + (size_t)l * HH;
    const float* l1b = ln1b + (size_t)l * HH;
    const float* l2w = ln2w + (size_t)l * HH;
    const float* l2b = ln2b + (size_t)l * HH;

    // per-layer weight convert+transpose to bf16 (L2-resident, 14 MB)
    cvt_wT<<<dim3(HH / 32, HH / 32), cb, 0, stream>>>(Wq_l, wqT, HH, HH);
    cvt_wT<<<dim3(HH / 32, HH / 32), cb, 0, stream>>>(Wk_l, wkT, HH, HH);
    cvt_wT<<<dim3(HH / 32, HH / 32), cb, 0, stream>>>(Wv_l, wvT, HH, HH);
    cvt_wT<<<dim3(HH / 32, HH / 32), cb, 0, stream>>>(Wo_l, woT, HH, HH);
    cvt_wT<<<dim3(FFF / 32, HH / 32), cb, 0, stream>>>(Wi_l, wiT, HH, FFF);
    cvt_wT<<<dim3(HH / 32, FFF / 32), cb, 0, stream>>>(Wf_l, wfT, FFF, HH);

    wmma_gemm<<<gH, 256, 0, stream>>>(hbf, wqT, bq_l, nullptr, nullptr, qb2,
                                      HH, HH, 0);
    wmma_gemm<<<gH, 256, 0, stream>>>(hbf, wkT, bk_l, nullptr, nullptr, kb2,
                                      HH, HH, 0);
    wmma_gemm<<<gH, 256, 0, stream>>>(hbf, wvT, bv_l, nullptr, nullptr, vb2,
                                      HH, HH, 0);
    flash_attn<<<gA, 128, 0, stream>>>(qb2, kb2, vb2, attn_mask, ctxb);
    wmma_gemm<<<gH, 256, 0, stream>>>(ctxb, woT, bo_l, h, tmp, nullptr,
                                      HH, HH, 1);
    layernorm_k<<<MTOT, 256, 0, stream>>>(tmp, l1w, l1b, attn, attnbf);
    wmma_gemm<<<gF, 256, 0, stream>>>(attnbf, wiT, bi_l, nullptr, nullptr, ffb,
                                      HH, FFF, 2);
    wmma_gemm<<<gH, 256, 0, stream>>>(ffb, wfT, bf_l, attn, tmp, nullptr,
                                      FFF, HH, 1);
    float* hout = (l == LL - 1) ? (float*)d_out : h;
    layernorm_k<<<MTOT, 256, 0, stream>>>(tmp, l2w, l2b, hout, hbf);
  }
}
